// RSSM_39994735460543
// MI455X (gfx1250) — compile-verified, matrix operand order checked by reading, not run on hardware
//
#include <hip/hip_runtime.h>

// ---------------------------------------------------------------------------
// Types for gfx1250 WMMA (wave32). bf16 A/B fragments, fp32 accumulator.
// ---------------------------------------------------------------------------
typedef __bf16 bf16t;
typedef __attribute__((ext_vector_type(16))) __bf16 v16bf;
typedef __attribute__((ext_vector_type(8)))  __bf16 v8bf;
typedef __attribute__((ext_vector_type(8)))  float  v8f;

#define TT 64
#define BB 1024
#define EMBED 1024
#define ACT 32
#define STOCH 32
#define DETER 1024
#define HID 1024

static __device__ __forceinline__ v16bf frag_cat(v8bf lo, v8bf hi) {
  return __builtin_shufflevector(lo, hi,
           0,1,2,3,4,5,6,7,8,9,10,11,12,13,14,15);
}

// ---------------------------------------------------------------------------
// bf16 WMMA GEMM: C[M,N](f32) = A[M,K](bf16,row-major) x Bt[N,K](bf16,row-major
// == B^T) + bias.  256 threads = 8 waves, mapped as WM x WN wave grid.
// Each wave computes a 16(M) x 64(N) strip = 4 x (16x16) WMMA tiles.
// Block tile: (WM*16) x (WN*64).  Grid must cover M,N exactly (no guards,
// EXEC stays all-1s around every WMMA as the ISA requires).
//
// Inner loop processes TWO K-steps (K is always a multiple of 64):
//   issue all 20 b128 fragment loads  ->  sched_barrier  ->  8 WMMAs.
// The sched_barrier stops the machine scheduler from sinking loads back next
// to their WMMAs (which would serialize load/wait/wmma with loadcnt 0x0); this
// keeps ~20 loads in flight and lets waits drain progressively.
//
// A/B fragment per-lane layout per CDNA5 ISA 7.12.2:
//   lane<16 : row (lane&15), K = {k0..k0+7, k0+16..k0+23}
//   lane>=16: row (lane&15), K = {k0+8..k0+15, k0+24..k0+31}
// ---------------------------------------------------------------------------
template <int WM, int WN>
__global__ __launch_bounds__(256) void wmma_gemm_bf16(
    const bf16t* __restrict__ A, const bf16t* __restrict__ Bt,
    const float* __restrict__ bias, float* __restrict__ C,
    int N, int K)
{
  const int lane = threadIdx.x & 31;
  const int wave = threadIdx.x >> 5;
  const int wm = wave % WM;
  const int wn = wave / WM;
  const int tileM = blockIdx.x * (WM * 16) + wm * 16;
  const int tileN = blockIdx.y * (WN * 64) + wn * 64;

  const int lrow = lane & 15;
  const int koff = (lane >> 4) * 8;              // 0 or 8

  const bf16t* aBase = A  + (size_t)(tileM + lrow) * K + koff;
  const bf16t* bBase = Bt + (size_t)(tileN + lrow) * K + koff;
  const bf16t* bp[4];
#pragma unroll
  for (int j = 0; j < 4; ++j) bp[j] = bBase + (size_t)j * 16 * K;

  v8f acc[4];
#pragma unroll
  for (int j = 0; j < 4; ++j) acc[j] = (v8f)(0.0f);

  for (int k0 = 0; k0 < K; k0 += 64) {
    // ---- issue ALL fragment loads for two K-steps ----
    v8bf a0lo = *(const v8bf*)(aBase + k0);
    v8bf a0hi = *(const v8bf*)(aBase + k0 + 16);
    v8bf a1lo = *(const v8bf*)(aBase + k0 + 32);
    v8bf a1hi = *(const v8bf*)(aBase + k0 + 48);
    v8bf b0lo[4], b0hi[4], b1lo[4], b1hi[4];
#pragma unroll
    for (int j = 0; j < 4; ++j) {
      b0lo[j] = *(const v8bf*)(bp[j] + k0);
      b0hi[j] = *(const v8bf*)(bp[j] + k0 + 16);
      b1lo[j] = *(const v8bf*)(bp[j] + k0 + 32);
      b1hi[j] = *(const v8bf*)(bp[j] + k0 + 48);
    }
    // Speculative prefetch of the next 2-step chunk (dropped past the end).
    __builtin_prefetch(aBase + k0 + 64, 0, 3);   // global_prefetch_b8
    __builtin_prefetch(bp[0] + k0 + 64, 0, 3);
    __builtin_prefetch(bp[2] + k0 + 64, 0, 3);

    // ---- nothing crosses this line: loads stay hoisted, WMMAs stay below ----
    __builtin_amdgcn_sched_barrier(0);

    const v16bf a0 = frag_cat(a0lo, a0hi);
#pragma unroll
    for (int j = 0; j < 4; ++j) {
      acc[j] = __builtin_amdgcn_wmma_f32_16x16x32_bf16(
          false, a0, false, frag_cat(b0lo[j], b0hi[j]), (short)0, acc[j],
          false, false);
    }
    const v16bf a1 = frag_cat(a1lo, a1hi);
#pragma unroll
    for (int j = 0; j < 4; ++j) {
      acc[j] = __builtin_amdgcn_wmma_f32_16x16x32_bf16(
          false, a1, false, frag_cat(b1lo[j], b1hi[j]), (short)0, acc[j],
          false, false);
    }
  }

  // Epilogue: C/D layout — lane<16: M=tileM+v; lane>=16: M=tileM+8+v; N=lane&15
  const int rbase = tileM + ((lane >> 4) ? 8 : 0);
#pragma unroll
  for (int j = 0; j < 4; ++j) {
    const int col = tileN + j * 16 + lrow;
    const float bv = bias[col];
#pragma unroll
    for (int v = 0; v < 8; ++v)
      C[(size_t)(rbase + v) * N + col] = acc[j][v] + bv;
  }
}

// ---------------------------------------------------------------------------
// LayerNorm (population var) + ELU, f32 in -> bf16 out. One block per row.
// ---------------------------------------------------------------------------
__global__ __launch_bounds__(256) void ln_elu_bf16(
    const float* __restrict__ X, const float* __restrict__ g,
    const float* __restrict__ be, bf16t* __restrict__ Y, int D)
{
  __shared__ float sh[16];
  const int row = blockIdx.x;
  const float* x = X + (size_t)row * D;
  float s = 0.f, s2 = 0.f;
  for (int i = threadIdx.x; i < D; i += 256) {
    float v = x[i]; s += v; s2 += v * v;
  }
  for (int off = 16; off; off >>= 1) {
    s  += __shfl_down(s,  off, 32);
    s2 += __shfl_down(s2, off, 32);
  }
  const int wv = threadIdx.x >> 5, ln = threadIdx.x & 31;
  if (ln == 0) { sh[wv] = s; sh[8 + wv] = s2; }
  __syncthreads();
  if (threadIdx.x == 0) {
    float ts = 0.f, ts2 = 0.f;
    for (int w = 0; w < 8; ++w) { ts += sh[w]; ts2 += sh[8 + w]; }
    const float m = ts / D;
    sh[0] = m;
    sh[1] = ts2 / D - m * m;
  }
  __syncthreads();
  const float mean = sh[0];
  const float inv  = rsqrtf(sh[1] + 1e-5f);
  bf16t* y = Y + (size_t)row * D;
  for (int i = threadIdx.x; i < D; i += 256) {
    float v = (x[i] - mean) * inv * g[i] + be[i];
    v = v > 0.f ? v : (expf(v) - 1.f);
    y[i] = (bf16t)v;
  }
}

// ---------------------------------------------------------------------------
// Per-step prep: mask stoch/deter by (1-is_first), build xin = [stoch|action].
// idx over B*1024.
// ---------------------------------------------------------------------------
__global__ __launch_bounds__(256) void prep_step(
    const float* __restrict__ stoch, const float* __restrict__ action_t,
    const float* __restrict__ isfirst_t, float* __restrict__ deterf,
    bf16t* __restrict__ deterb, bf16t* __restrict__ xin)
{
  const int idx = blockIdx.x * 256 + threadIdx.x;
  const int b = idx >> 10, j = idx & 1023;
  const float mask = 1.f - isfirst_t[b];
  const float d = deterf[idx] * mask;
  deterf[idx] = d;
  deterb[idx] = (bf16t)d;
  if (j < STOCH + ACT) {
    const float v = (j < STOCH) ? stoch[b * STOCH + j] * mask
                                : action_t[b * ACT + (j - STOCH)] * mask;
    xin[b * (STOCH + ACT) + j] = (bf16t)v;
  }
}

// ---------------------------------------------------------------------------
// GRU gate fusion + deter update; writes deter slice of feats. idx over B*1024.
// ---------------------------------------------------------------------------
__global__ __launch_bounds__(256) void gru_update(
    const float* __restrict__ gi, const float* __restrict__ gh,
    float* __restrict__ deterf, bf16t* __restrict__ deterb,
    float* __restrict__ featT)
{
  const int idx = blockIdx.x * 256 + threadIdx.x;
  const int b = idx >> 10, j = idx & 1023;
  const size_t base = (size_t)b * (3 * DETER);
  const float ir = gi[base + j], iz = gi[base + DETER + j], in = gi[base + 2 * DETER + j];
  const float hr = gh[base + j], hz = gh[base + DETER + j], hn = gh[base + 2 * DETER + j];
  const float r = 1.f / (1.f + expf(-(ir + hr)));
  const float z = 1.f / (1.f + expf(-(iz + hz)));
  const float n = tanhf(in + r * hn);
  float d = deterf[idx];
  d = (1.f - z) * n + z * d;
  deterf[idx] = d;
  deterb[idx] = (bf16t)d;
  featT[(size_t)b * (STOCH + DETER) + STOCH + j] = d;
}

// qin = bf16([deter | embed_t]); idx over B*2048
__global__ __launch_bounds__(256) void prep_qin(
    const float* __restrict__ deterf, const float* __restrict__ embed_t,
    bf16t* __restrict__ qin)
{
  const int idx = blockIdx.x * 256 + threadIdx.x;
  const int b = idx >> 11, j = idx & 2047;
  const float v = (j < DETER) ? deterf[b * DETER + j]
                              : embed_t[b * EMBED + (j - DETER)];
  qin[idx] = (bf16t)v;
}

// prior head epilogue; idx over B*32
__global__ __launch_bounds__(256) void head_p(
    const float* __restrict__ pp, float* __restrict__ pm, float* __restrict__ ps)
{
  const int idx = blockIdx.x * 256 + threadIdx.x;
  const int b = idx >> 5, s = idx & 31;
  pm[idx] = pp[b * 64 + s];
  const float raw = pp[b * 64 + 32 + s];
  ps[idx] = 2.f / (1.f + expf(-0.5f * raw)) + 0.1f;
}

// posterior head epilogue; also updates stoch state and feats[:, :32]
__global__ __launch_bounds__(256) void head_q(
    const float* __restrict__ qq, float* __restrict__ qm, float* __restrict__ qs,
    float* __restrict__ stoch, float* __restrict__ featT)
{
  const int idx = blockIdx.x * 256 + threadIdx.x;
  const int b = idx >> 5, s = idx & 31;
  const float m = qq[b * 64 + s];
  const float raw = qq[b * 64 + 32 + s];
  qm[idx] = m;
  qs[idx] = 2.f / (1.f + expf(-0.5f * raw)) + 0.1f;
  stoch[idx] = m;
  featT[(size_t)b * (STOCH + DETER) + s] = m;
}

// W[K,N] f32 -> Wt[N,K] bf16 (transpose + downconvert); idx over K*N
__global__ __launch_bounds__(256) void w_to_bf16_t(
    const float* __restrict__ W, bf16t* __restrict__ Wt, int K, int N)
{
  const int idx = blockIdx.x * 256 + threadIdx.x;
  if (idx >= K * N) return;
  const int n = idx / K, k = idx - n * K;
  Wt[idx] = (bf16t)W[(size_t)k * N + n];
}

// zero initial state; idx over B*1024
__global__ __launch_bounds__(256) void zero_state(
    float* __restrict__ stoch, float* __restrict__ deterf, bf16t* __restrict__ deterb)
{
  const int idx = blockIdx.x * 256 + threadIdx.x;
  if (idx < BB * STOCH) stoch[idx] = 0.f;
  deterf[idx] = 0.f;
  deterb[idx] = (bf16t)0.f;
}

// ---------------------------------------------------------------------------
extern "C" void kernel_launch(void* const* d_in, const int* in_sizes, int n_in,
                              void* d_out, int out_size, void* d_ws, size_t ws_size,
                              hipStream_t stream) {
  (void)in_sizes; (void)n_in; (void)out_size; (void)ws_size;

  const float* embed   = (const float*)d_in[0];   // [T,B,1024]
  const float* action  = (const float*)d_in[1];   // [T,B,32]
  const float* isfirst = (const float*)d_in[2];   // [T,B]
  const float* W_pre = (const float*)d_in[3];
  const float* b_pre = (const float*)d_in[4];
  const float* g_pre = (const float*)d_in[5];
  const float* be_pre= (const float*)d_in[6];
  const float* Wi    = (const float*)d_in[7];
  const float* Wh    = (const float*)d_in[8];
  const float* bi    = (const float*)d_in[9];
  const float* bh    = (const float*)d_in[10];
  const float* Wp1   = (const float*)d_in[11];
  const float* bp1   = (const float*)d_in[12];
  const float* g_p   = (const float*)d_in[13];
  const float* be_p  = (const float*)d_in[14];
  const float* Wp2   = (const float*)d_in[15];
  const float* bp2   = (const float*)d_in[16];
  const float* Wq1   = (const float*)d_in[17];
  const float* bq1   = (const float*)d_in[18];
  const float* g_q   = (const float*)d_in[19];
  const float* be_q  = (const float*)d_in[20];
  const float* Wq2   = (const float*)d_in[21];
  const float* bq2   = (const float*)d_in[22];

  float* out = (float*)d_out;
  const size_t PM_OFF = 0;
  const size_t PS_OFF = (size_t)TT * BB * STOCH;
  const size_t QM_OFF = 2 * PS_OFF;
  const size_t QS_OFF = 3 * PS_OFF;
  const size_t FT_OFF = 4 * PS_OFF;

  // Workspace carve-up (256B aligned)
  char* ws = (char*)d_ws;
  size_t off = 0;
  auto carve = [&](size_t bytes) -> void* {
    void* p = ws + off;
    off = (off + bytes + 255) & ~(size_t)255;
    return p;
  };
  bf16t* WpreT = (bf16t*)carve((size_t)HID * 64 * 2);
  bf16t* WiT   = (bf16t*)carve((size_t)3 * DETER * HID * 2);
  bf16t* WhT   = (bf16t*)carve((size_t)3 * DETER * DETER * 2);
  bf16t* Wp1T  = (bf16t*)carve((size_t)HID * DETER * 2);
  bf16t* Wp2T  = (bf16t*)carve((size_t)64 * HID * 2);
  bf16t* Wq1T  = (bf16t*)carve((size_t)HID * (DETER + EMBED) * 2);
  bf16t* Wq2T  = (bf16t*)carve((size_t)64 * HID * 2);
  bf16t* xin   = (bf16t*)carve((size_t)BB * 64 * 2);
  bf16t* xbf   = (bf16t*)carve((size_t)BB * HID * 2);
  bf16t* hbf   = (bf16t*)carve((size_t)BB * HID * 2);
  bf16t* qin   = (bf16t*)carve((size_t)BB * (DETER + EMBED) * 2);
  float* deterf= (float*)carve((size_t)BB * DETER * 4);
  bf16t* deterb= (bf16t*)carve((size_t)BB * DETER * 2);
  float* stochf= (float*)carve((size_t)BB * STOCH * 4);
  float* gi_b  = (float*)carve((size_t)BB * 3 * DETER * 4);
  float* gh_b  = (float*)carve((size_t)BB * 3 * DETER * 4);
  float* tmp   = (float*)carve((size_t)BB * HID * 4);
  float* pp    = (float*)carve((size_t)BB * 64 * 4);
  float* qq    = (float*)carve((size_t)BB * 64 * 4);

  // One-time (per launch) weight convert+transpose, state init
  auto cvt = [&](const float* W, bf16t* Wt, int K, int N) {
    const int n = K * N;
    w_to_bf16_t<<<(n + 255) / 256, 256, 0, stream>>>(W, Wt, K, N);
  };
  cvt(W_pre, WpreT, 64, HID);
  cvt(Wi,    WiT,   HID, 3 * DETER);
  cvt(Wh,    WhT,   DETER, 3 * DETER);
  cvt(Wp1,   Wp1T,  DETER, HID);
  cvt(Wp2,   Wp2T,  HID, 64);
  cvt(Wq1,   Wq1T,  DETER + EMBED, HID);
  cvt(Wq2,   Wq2T,  HID, 64);
  zero_state<<<(BB * DETER) / 256, 256, 0, stream>>>(stochf, deterf, deterb);

  // Wave-grid <4,2>: block tile 64x128.  Wave-grid <8,1>: block tile 128x64.
  const dim3 gN1024(BB / 64, HID / 128);          // 16 x 8
  const dim3 gN3072(BB / 64, (3 * DETER) / 128);  // 16 x 24
  const dim3 gN64  (BB / 128, 1);                 // 8 x 1

  for (int t = 0; t < TT; ++t) {
    const float* action_t = action + (size_t)t * BB * ACT;
    const float* embed_t  = embed  + (size_t)t * BB * EMBED;
    const float* if_t     = isfirst + (size_t)t * BB;
    float* featT = out + FT_OFF + (size_t)t * BB * (STOCH + DETER);

    // mask state, build xin = [stoch|action]
    prep_step<<<(BB * DETER) / 256, 256, 0, stream>>>(
        stochf, action_t, if_t, deterf, deterb, xin);

    // pre-GRU MLP: xin @ W_pre -> LN -> ELU -> x (bf16)
    wmma_gemm_bf16<4, 2><<<gN1024, 256, 0, stream>>>(xin, WpreT, b_pre, tmp, HID, 64);
    ln_elu_bf16<<<BB, 256, 0, stream>>>(tmp, g_pre, be_pre, xbf, HID);

    // GRU gates
    wmma_gemm_bf16<4, 2><<<gN3072, 256, 0, stream>>>(xbf,    WiT, bi, gi_b, 3 * DETER, HID);
    wmma_gemm_bf16<4, 2><<<gN3072, 256, 0, stream>>>(deterb, WhT, bh, gh_b, 3 * DETER, DETER);
    gru_update<<<(BB * DETER) / 256, 256, 0, stream>>>(gi_b, gh_b, deterf, deterb, featT);

    // prior head
    wmma_gemm_bf16<4, 2><<<gN1024, 256, 0, stream>>>(deterb, Wp1T, bp1, tmp, HID, DETER);
    ln_elu_bf16<<<BB, 256, 0, stream>>>(tmp, g_p, be_p, hbf, HID);
    wmma_gemm_bf16<8, 1><<<gN64, 256, 0, stream>>>(hbf, Wp2T, bp2, pp, 64, HID);
    head_p<<<(BB * STOCH) / 256, 256, 0, stream>>>(
        pp, out + PM_OFF + (size_t)t * BB * STOCH, out + PS_OFF + (size_t)t * BB * STOCH);

    // posterior head
    prep_qin<<<(BB * (DETER + EMBED)) / 256, 256, 0, stream>>>(deterf, embed_t, qin);
    wmma_gemm_bf16<4, 2><<<gN1024, 256, 0, stream>>>(qin, Wq1T, bq1, tmp, HID, DETER + EMBED);
    ln_elu_bf16<<<BB, 256, 0, stream>>>(tmp, g_q, be_q, hbf, HID);
    wmma_gemm_bf16<8, 1><<<gN64, 256, 0, stream>>>(hbf, Wq2T, bq2, qq, 64, HID);
    head_q<<<(BB * STOCH) / 256, 256, 0, stream>>>(
        qq, out + QM_OFF + (size_t)t * BB * STOCH, out + QS_OFF + (size_t)t * BB * STOCH,
        stochf, featT);
  }
}
